// Syntax_Transformer_RNNModel_15779709845999
// MI455X (gfx1250) — matrix operand order repaired and checked by reading, not compiled, not run on hardware
//
#include <hip/hip_runtime.h>
#include <math.h>

#define B_   8
#define S_   128
#define DE_  128
#define HID_ 512
#define NH_  8
#define HDE_ 16
#define HDT_ 64
#define L_   2
#define NEG_ (-10000.0f)
#define EPS_ 1e-5f
#define WE_  0.5f

typedef __bf16 bf16;
typedef __attribute__((ext_vector_type(16))) __bf16 v16bf;
typedef __attribute__((ext_vector_type(8)))  __bf16 v8bf;
typedef __attribute__((ext_vector_type(4)))  __bf16 v4bf;
typedef __attribute__((ext_vector_type(8)))  float v8f;

__device__ __forceinline__ v16bf cat8(v8bf lo, v8bf hi) {
  return __builtin_shufflevector(lo, hi, 0,1,2,3,4,5,6,7,8,9,10,11,12,13,14,15);
}

// A-operand (16-bit, 16x32): lane owns row (lane&15);
// elements 0..7 -> K = 8*(lane>=16)+0..7 ; elements 8..15 -> +16.
// base = &row[k0]; both halves are 16B-aligned contiguous 8xbf16.
__device__ __forceinline__ v16bf loadA(const bf16* base, int lane) {
  const bf16* p = base + ((lane >> 4) << 3);
  v8bf lo = *(const v8bf*)p;
  v8bf hi = *(const v8bf*)(p + 16);
  return cat8(lo, hi);
}
// B-operand (32x16): lane owns col (lane&15); elements 0..7 -> K = 8*(lane>=16)+0..7 ; elements 8..15 -> +16.
// base = &colT[k0] in a [n][k] transposed LDS image.
__device__ __forceinline__ v16bf loadB(const bf16* base, int lane) {
  const bf16* p = base + ((lane >> 4) << 3);
  v8bf lo = *(const v8bf*)p;
  v8bf hi = *(const v8bf*)(p + 16);
  return cat8(lo, hi);
}

__device__ __forceinline__ v8f wmma_bf16(v16bf a, v16bf b, v8f c) {
  return __builtin_amdgcn_wmma_f32_16x16x32_bf16(false, a, false, b,
                                                 (short)0, c, false, false);
}

// (128x128 slab) @ (128x16 weight slice, transposed LDS [n][k]) + bias.
// Output stored to LDS at outp[row*orow + n*ocol] as bf16.
__device__ __forceinline__ void proj_tile(const bf16* sE, const bf16* swT,
                                          float bs, bf16* outp,
                                          int orow, int ocol,
                                          int w, int lane) {
  const int n_   = lane & 15;
  const int rowA = (w << 4) + n_;
  v8f c = {0.f,0.f,0.f,0.f,0.f,0.f,0.f,0.f};
  for (int kk = 0; kk < 4; ++kk) {
    int k0 = kk << 5;
    v16bf a  = loadA(sE + rowA * DE_ + k0, lane);
    v16bf bb = loadB(swT + n_ * DE_ + k0, lane);
    c = wmma_bf16(a, bb, c);
  }
  #pragma unroll
  for (int r = 0; r < 8; ++r) {
    int row = (w << 4) + ((lane >> 4) << 3) + r;
    outp[row * orow + n_ * ocol] = (bf16)(c[r] + bs);
  }
}

// ---------------------------------------------------------------------------
// Kernel 1: dynamic edge-attention. One block per (b, i, h).
// ---------------------------------------------------------------------------
__global__ __launch_bounds__(256) void k_dyn_attn(
    const float* __restrict__ edge, const int* __restrict__ masks,
    const float* __restrict__ wq, const float* __restrict__ bq,
    const float* __restrict__ wk, const float* __restrict__ bk,
    const float* __restrict__ wv, const float* __restrict__ bv,
    float* __restrict__ merged)
{
  __shared__ bf16 sE[S_ * DE_];      // 32KB edge slab; reused as probs sP
  __shared__ bf16 sQ[S_ * HDE_];     // [row][d]
  __shared__ bf16 sK[S_ * HDE_];     // [row][d]
  __shared__ bf16 sVt[HDE_ * S_];    // [n][k] transposed (B operand of P@V)
  __shared__ bf16 sWqT[HDE_ * DE_];  // [n][k] transposed weight slices
  __shared__ bf16 sWkT[HDE_ * DE_];
  __shared__ bf16 sWvT[HDE_ * DE_];

  const int tid  = threadIdx.x;
  const int lane = tid & 31;
  const int w    = tid >> 5;
  const int blk  = blockIdx.x;
  const int h    = blk & (NH_ - 1);
  const int bi   = blk >> 3;          // b*S + i

  const size_t ebase = (size_t)bi * S_ * DE_;
  #pragma unroll
  for (int it = 0; it < 16; ++it) {           // 16384 f32 -> bf16, vectorized
    int elt = (tid + it * 256) * 4;
    float4 f = *(const float4*)(edge + ebase + elt);
    v4bf o = {(bf16)f.x, (bf16)f.y, (bf16)f.z, (bf16)f.w};
    *(v4bf*)(sE + elt) = o;
  }
  for (int idx = tid; idx < HDE_ * DE_; idx += 256) {
    int n = idx >> 7, k = idx & 127;          // [n][k]
    int col = h * HDE_ + n;
    sWqT[idx] = (bf16)wq[k * DE_ + col];
    sWkT[idx] = (bf16)wk[k * DE_ + col];
    sWvT[idx] = (bf16)wv[k * DE_ + col];
  }
  __syncthreads();

  const int n_   = lane & 15;
  const int rowA = (w << 4) + n_;

  // ---- projections: wave w computes rows [16w,16w+16) of q_h/k_h/v_h ----
  proj_tile(sE, sWqT, bq[h * HDE_ + n_], sQ,  HDE_, 1, w, lane);
  proj_tile(sE, sWkT, bk[h * HDE_ + n_], sK,  HDE_, 1, w, lane);
  proj_tile(sE, sWvT, bv[h * HDE_ + n_], sVt, 1,  S_, w, lane);   // transposed
  __syncthreads();   // sE dead from here; alias as probs buffer
  bf16* sP = sE;

  // ---- scores (j-tile w, all 8 k-tiles), mask, row softmax, *0.25 ----
  {
    const v8bf vz = {(bf16)0.f,(bf16)0.f,(bf16)0.f,(bf16)0.f,
                     (bf16)0.f,(bf16)0.f,(bf16)0.f,(bf16)0.f};
    // A fragment: q rows, K=d (0..15 real, 16..31 zero-pad)
    v8bf alo = *(const v8bf*)(sQ + rowA * HDE_ + ((lane >> 4) << 3));
    v16bf a = cat8(alo, vz);
    float acc[8][8];
    for (int kt = 0; kt < 8; ++kt) {
      // B fragment: column k = kt*16+n_, K=d; elements 0..7 -> d = 8*(lane>=16)+0..7, elements 8..15 (K>=16) -> zero
      const bf16* krow = sK + ((kt << 4) + n_) * HDE_;
      v8bf blo = *(const v8bf*)(krow + ((lane >> 4) << 3));
      v8bf bhi = vz;
      v16bf bb = cat8(blo, bhi);
      v8f c = {0.f,0.f,0.f,0.f,0.f,0.f,0.f,0.f};
      c = wmma_bf16(a, bb, c);
      #pragma unroll
      for (int r = 0; r < 8; ++r) acc[kt][r] = c[r];
    }
    #pragma unroll
    for (int r = 0; r < 8; ++r) {
      int j = (w << 4) + ((lane >> 4) << 3) + r;
      if (masks[(size_t)bi * S_ + j] == 0)
        for (int kt = 0; kt < 8; ++kt) acc[kt][r] = NEG_;
    }
    #pragma unroll
    for (int r = 0; r < 8; ++r) {
      float mx = -3.4e38f;
      #pragma unroll
      for (int kt = 0; kt < 8; ++kt) mx = fmaxf(mx, acc[kt][r]);
      mx = fmaxf(mx, __shfl_xor(mx, 1, 32));
      mx = fmaxf(mx, __shfl_xor(mx, 2, 32));
      mx = fmaxf(mx, __shfl_xor(mx, 4, 32));
      mx = fmaxf(mx, __shfl_xor(mx, 8, 32));
      float ex[8]; float sum = 0.f;
      #pragma unroll
      for (int kt = 0; kt < 8; ++kt) { ex[kt] = __expf(acc[kt][r] - mx); sum += ex[kt]; }
      sum += __shfl_xor(sum, 1, 32);
      sum += __shfl_xor(sum, 2, 32);
      sum += __shfl_xor(sum, 4, 32);
      sum += __shfl_xor(sum, 8, 32);
      float inv = 0.25f / sum;   // softmax / sqrt(HD_E=16)
      int j = (w << 4) + ((lane >> 4) << 3) + r;
      #pragma unroll
      for (int kt = 0; kt < 8; ++kt)
        sP[j * S_ + (kt << 4) + n_] = (bf16)(ex[kt] * inv);
    }
  }
  __syncthreads();

  // ---- merged_h = P (128x128) @ v_h (128x16, LDS image transposed) ----
  {
    v8f c = {0.f,0.f,0.f,0.f,0.f,0.f,0.f,0.f};
    for (int kk = 0; kk < 4; ++kk) {
      int k0 = kk << 5;
      v16bf a  = loadA(sP + rowA * S_ + k0, lane);
      v16bf bb = loadB(sVt + n_ * S_ + k0, lane);
      c = wmma_bf16(a, bb, c);
    }
    #pragma unroll
    for (int r = 0; r < 8; ++r) {
      int j = (w << 4) + ((lane >> 4) << 3) + r;
      merged[((size_t)bi * S_ + j) * DE_ + h * HDE_ + n_] = c[r];
    }
  }
}

// ---------------------------------------------------------------------------
// Kernel 2: alpha gate + transpose mix. One wave per (b,i,j).
// ---------------------------------------------------------------------------
__global__ __launch_bounds__(256) void k_alpha_mix(
    const float* __restrict__ merged, const float* __restrict__ aw,
    const float* __restrict__ ab, bf16* __restrict__ edge_feat)
{
  int gw   = blockIdx.x * 8 + (threadIdx.x >> 5);
  int lane = threadIdx.x & 31;
  int b = gw >> 14;
  int i = (gw >> 7) & 127;
  int j = gw & 127;
  size_t o1 = (((size_t)b * S_ + i) * S_ + j) * DE_;
  size_t o2 = (((size_t)b * S_ + j) * S_ + i) * DE_;
  float m1[4], m2[4], part = 0.f;
  #pragma unroll
  for (int t = 0; t < 4; ++t) {
    int d = lane + (t << 5);
    m1[t] = merged[o1 + d];
    m2[t] = merged[o2 + d];
    part += m1[t] * aw[d] + m2[t] * aw[DE_ + d];
  }
  part += __shfl_xor(part, 1, 32);
  part += __shfl_xor(part, 2, 32);
  part += __shfl_xor(part, 4, 32);
  part += __shfl_xor(part, 8, 32);
  part += __shfl_xor(part, 16, 32);
  float al = 1.f / (1.f + __expf(-(part + ab[0])));
  #pragma unroll
  for (int t = 0; t < 4; ++t) {
    int d = lane + (t << 5);
    edge_feat[o1 + d] = (bf16)((1.f - al) * m1[t] + al * m2[t]);
  }
}

// ---------------------------------------------------------------------------
// Kernel 3: Q/K/V projection, WMMA GEMM 1024x512x512 (x3 weights).
// grid (64, 4): block = 16 rows x 128 cols, 8 waves -> one N-tile each.
// ---------------------------------------------------------------------------
__global__ __launch_bounds__(256) void k_qkv(
    const float* __restrict__ x,
    const float* __restrict__ Wq, const float* __restrict__ bq,
    const float* __restrict__ Wk, const float* __restrict__ bk,
    const float* __restrict__ Wv, const float* __restrict__ bv,
    float* __restrict__ Q, float* __restrict__ K, float* __restrict__ V)
{
  __shared__ bf16 sX[16 * HID_];        // 16KB [row][k]
  __shared__ bf16 sWs[3 * 128 * 32];    // 24KB [m][c][k] (transposed K-slice)
  const int tid  = threadIdx.x;
  const int lane = tid & 31;
  const int w    = tid >> 5;
  const int row0 = blockIdx.x << 4;
  const int col0 = blockIdx.y << 7;
  const int n_   = lane & 15;

  #pragma unroll
  for (int it = 0; it < 8; ++it) {           // 8192 f32 -> bf16, vectorized
    int elt = (tid + it * 256) * 4;
    float4 f = *(const float4*)(x + (size_t)row0 * HID_ + elt);
    v4bf o = {(bf16)f.x, (bf16)f.y, (bf16)f.z, (bf16)f.w};
    *(v4bf*)(sX + elt) = o;
  }

  const float* Wp[3] = {Wq, Wk, Wv};    // runtime (generic) pointers: OK
  v8f zero = {0.f,0.f,0.f,0.f,0.f,0.f,0.f,0.f};
  v8f c[3] = {zero, zero, zero};

  for (int kk = 0; kk < 16; ++kk) {
    int k0 = kk << 5;
    __syncthreads();
    for (int idx = tid; idx < 3 * 128 * 32; idx += 256) {
      int m = idx >> 12, rem = idx & 4095;
      int k = rem >> 7,  cc  = rem & 127;     // global read coalesced over cc
      sWs[((m << 7) + cc) * 32 + k] = (bf16)Wp[m][(size_t)(k0 + k) * HID_ + col0 + cc];
    }
    __syncthreads();
    v16bf a = loadA(sX + n_ * HID_ + k0, lane);
    #pragma unroll
    for (int m = 0; m < 3; ++m) {
      v16bf bb = loadB(sWs + ((m << 7) + (w << 4) + n_) * 32, lane);
      c[m] = wmma_bf16(a, bb, c[m]);
    }
  }
  const float* bp[3] = {bq, bk, bv};
  float* op[3] = {Q, K, V};
  int col = col0 + (w << 4) + n_;
  #pragma unroll
  for (int m = 0; m < 3; ++m) {
    float bs = bp[m][col];
    #pragma unroll
    for (int r = 0; r < 8; ++r) {
      int row = row0 + ((lane >> 4) << 3) + r;
      op[m][(size_t)row * HID_ + col] = c[m][r] + bs;
    }
  }
}

// ---------------------------------------------------------------------------
// Kernel 4: fused syntax attention + residual + LayerNorm. Block per (b,i).
// q.ek[j] = F[j,:].(ekw_h q_h) + q_h.ekb_h ; sum_j p_j ev[j] = (p^T F) evw_h + evb_h
// Edge-feature slab staged with CDNA5 async global->LDS copy (ASYNCcnt).
// ---------------------------------------------------------------------------
__global__ __launch_bounds__(256) void k_syn_attn(
    const float* __restrict__ x, const bf16* __restrict__ edge_feat,
    const int* __restrict__ masks,
    const float* __restrict__ Q, const float* __restrict__ Kb,
    const float* __restrict__ Vb,
    const float* __restrict__ ekw, const float* __restrict__ ekb,
    const float* __restrict__ evw, const float* __restrict__ evb,
    const float* __restrict__ g, const float* __restrict__ beta,
    float* __restrict__ y)
{
  __shared__ bf16  sF[S_ * DE_];   // 32KB edge-feature slab (bf16)
  __shared__ float sq[HID_];
  __shared__ float su[S_];
  __shared__ float st[S_];
  __shared__ float sS[S_];
  __shared__ float sPb[S_];
  __shared__ float sCtx[HID_];
  __shared__ float sY[HID_];

  const int tid = threadIdx.x;
  const int bi  = blockIdx.x;      // b*S + i
  const int b   = bi >> 7;

  // async global->LDS copy of the 32KB bf16 slab (16B per lane per issue)
  {
    const bf16* gsrc = edge_feat + (size_t)bi * S_ * DE_;
    #pragma unroll
    for (int it = 0; it < 8; ++it) {
      int elt = (tid + it * 256) * 8;
      unsigned lds_off = (unsigned)(uintptr_t)(sF + elt);
      const bf16* gp = gsrc + elt;
      asm volatile("global_load_async_to_lds_b128 %0, %1, off"
                   :: "v"(lds_off), "v"(gp) : "memory");
    }
  }
  for (int idx = tid; idx < HID_; idx += 256)
    sq[idx] = Q[(size_t)bi * HID_ + idx];
  asm volatile("s_wait_asynccnt 0x0" ::: "memory");
  __syncthreads();

  for (int h = 0; h < NH_; ++h) {
    const int hb = h * HDT_;
    if (tid < S_) {                               // u = ekw_h . q_h
      float u = 0.f;
      for (int d = 0; d < HDT_; ++d)
        u += ekw[(size_t)tid * HID_ + hb + d] * sq[hb + d];
      su[tid] = u;
    }
    __syncthreads();
    if (tid < S_) {                               // scores
      const int j = tid;
      float qb = 0.f, qk = 0.f;
      const float* krow = Kb + (size_t)(b * S_ + j) * HID_ + hb;
      for (int d = 0; d < HDT_; ++d) {
        qb += sq[hb + d] * ekb[hb + d];
        qk += sq[hb + d] * krow[d];
      }
      float qek = qb;
      for (int c = 0; c < DE_; ++c)
        qek += (float)sF[j * DE_ + c] * su[c];
      float s = (qk + WE_ * qek) * 0.125f;        // 1/sqrt(HD_T=64)
      if (masks[(size_t)bi * S_ + j] == 0) s = NEG_;
      sS[j] = s;
    }
    __syncthreads();
    if (tid < S_) {                               // softmax over j
      float mx = -3.4e38f;
      for (int j2 = 0; j2 < S_; ++j2) mx = fmaxf(mx, sS[j2]);
      float sum = 0.f;
      for (int j2 = 0; j2 < S_; ++j2) sum += __expf(sS[j2] - mx);
      sPb[tid] = __expf(sS[tid] - mx) / sum;
    }
    __syncthreads();
    if (tid < S_) {                               // t = p^T F
      float t = 0.f;
      for (int j2 = 0; j2 < S_; ++j2)
        t += sPb[j2] * (float)sF[j2 * DE_ + tid];
      st[tid] = t;
    }
    __syncthreads();
    if (tid < HDT_) {                             // ctx_h
      const int d = tid;
      float accV = 0.f;
      for (int j2 = 0; j2 < S_; ++j2)
        accV += sPb[j2] * Vb[(size_t)(b * S_ + j2) * HID_ + hb + d];
      float accE = 0.f;
      for (int c = 0; c < DE_; ++c)
        accE += st[c] * evw[(size_t)c * HID_ + hb + d];
      sCtx[hb + d] = accV + WE_ * (accE + evb[hb + d]);
    }
    __syncthreads();
  }

  // residual + LayerNorm
  for (int idx = tid; idx < HID_; idx += 256)
    sY[idx] = x[(size_t)bi * HID_ + idx] + sCtx[idx];
  __syncthreads();
  float mu = 0.f;
  for (int c = 0; c < HID_; ++c) mu += sY[c];
  mu *= (1.f / HID_);
  float var = 0.f;
  for (int c = 0; c < HID_; ++c) { float d = sY[c] - mu; var += d * d; }
  var *= (1.f / HID_);
  float inv = rsqrtf(var + EPS_);
  for (int idx = tid; idx < HID_; idx += 256)
    y[(size_t)bi * HID_ + idx] = (sY[idx] - mu) * inv * g[idx] + beta[idx];
}

// ---------------------------------------------------------------------------
extern "C" void kernel_launch(void* const* d_in, const int* in_sizes, int n_in,
                              void* d_out, int out_size, void* d_ws, size_t ws_size,
                              hipStream_t stream)
{
  const float* edge  = (const float*)d_in[0];
  const float* tok   = (const float*)d_in[1];
  const int*   masks = (const int*)  d_in[2];
  const float* dl_wq = (const float*)d_in[3];
  const float* dl_bq = (const float*)d_in[4];
  const float* dl_wk = (const float*)d_in[5];
  const float* dl_bk = (const float*)d_in[6];
  const float* dl_wv = (const float*)d_in[7];
  const float* dl_bv = (const float*)d_in[8];
  const float* aw    = (const float*)d_in[9];
  const float* ab    = (const float*)d_in[10];
  const float* Wq    = (const float*)d_in[11];
  const float* bq    = (const float*)d_in[12];
  const float* Wk    = (const float*)d_in[13];
  const float* bk    = (const float*)d_in[14];
  const float* Wv    = (const float*)d_in[15];
  const float* bv    = (const float*)d_in[16];
  const float* ekw   = (const float*)d_in[17];
  const float* ekb   = (const float*)d_in[18];
  const float* evw   = (const float*)d_in[19];
  const float* evb   = (const float*)d_in[20];
  const float* lng   = (const float*)d_in[21];
  const float* lnb   = (const float*)d_in[22];

  char* ws = (char*)d_ws;
  const size_t mergedBytes = (size_t)B_ * S_ * S_ * DE_ * sizeof(float);   // 64MB
  const size_t efeatBytes  = (size_t)B_ * S_ * S_ * DE_ * sizeof(bf16);    // 32MB
  float* merged = (float*)ws;
  bf16*  efeat  = (bf16*)(ws + mergedBytes);
  float* Qb     = (float*)(ws + mergedBytes + efeatBytes);
  float* Kbuf   = Qb   + (size_t)B_ * S_ * HID_;
  float* Vbuf   = Kbuf + (size_t)B_ * S_ * HID_;
  float* xbuf   = Vbuf + (size_t)B_ * S_ * HID_;

  // dynamic edge layer
  k_dyn_attn<<<B_ * S_ * NH_, 256, 0, stream>>>(
      edge, masks, dl_wq, dl_bq, dl_wk, dl_bk, dl_wv, dl_bv, merged);
  k_alpha_mix<<<(B_ * S_ * S_) / 8, 256, 0, stream>>>(merged, aw, ab, efeat);

  // syntax layers
  const float* xs = tok;
  for (int l = 0; l < L_; ++l) {
    k_qkv<<<dim3(64, 4), 256, 0, stream>>>(
        xs,
        Wq + (size_t)l * HID_ * HID_, bq + (size_t)l * HID_,
        Wk + (size_t)l * HID_ * HID_, bk + (size_t)l * HID_,
        Wv + (size_t)l * HID_ * HID_, bv + (size_t)l * HID_,
        Qb, Kbuf, Vbuf);
    float* yd = (l == L_ - 1) ? (float*)d_out : xbuf;
    k_syn_attn<<<B_ * S_, 256, 0, stream>>>(
        xs, efeat, masks, Qb, Kbuf, Vbuf,
        ekw + (size_t)l * DE_ * HID_, ekb + (size_t)l * HID_,
        evw + (size_t)l * DE_ * HID_, evb + (size_t)l * HID_,
        lng + (size_t)l * HID_, lnb + (size_t)l * HID_, yd);
    xs = yd;
  }
}
